// MultiHeadAttention_5506148073999
// MI455X (gfx1250) — matrix operand order repaired and checked by reading, not compiled, not run on hardware
//
#include <hip/hip_runtime.h>
#include <hip/hip_bf16.h>
#include <math.h>

#define D_MODEL 1024
#define NHEADS  16
#define DK      64
#define BATCH   4
#define SEQ     1024
#define MROWS   (BATCH * SEQ)      // 4096 token rows

typedef __bf16 bf16_t;
typedef __attribute__((ext_vector_type(16))) __bf16 v16bf;
typedef __attribute__((ext_vector_type(8)))  __bf16 v8bf;
typedef __attribute__((ext_vector_type(8)))  float  v8f;
typedef __attribute__((ext_vector_type(4)))  float  f32x4;

static __device__ __forceinline__ v16bf cat8(v8bf lo, v8bf hi) {
  return __builtin_shufflevector(lo, hi, 0,1,2,3,4,5,6,7,8,9,10,11,12,13,14,15);
}

static __device__ __forceinline__ v8bf cvt8(f32x4 a, f32x4 b) {
  v8bf r;
  r[0] = (__bf16)a[0]; r[1] = (__bf16)a[1]; r[2] = (__bf16)a[2]; r[3] = (__bf16)a[3];
  r[4] = (__bf16)b[0]; r[5] = (__bf16)b[1]; r[6] = (__bf16)b[2]; r[7] = (__bf16)b[3];
  return r;
}

// A-fragment (16x32 bf16, contiguous k order) from a bf16 row-major row pointer.
// Lane: row = lane&15, hi = lane>>4. Halves 0-15 -> K = hi*16 + 0..15.
static __device__ __forceinline__ v16bf load_a_bf16(const bf16_t* __restrict__ rowp, int hi) {
  v8bf lo = *(const v8bf*)(rowp + hi * 16);
  v8bf hb = *(const v8bf*)(rowp + hi * 16 + 8);
  return cat8(lo, hb);
}

// B-fragment (32x16 bf16, contiguous k order): halves 0-15 -> K = hi*16 + 0..15 (one 32B load).
static __device__ __forceinline__ v16bf load_b_bf16(const bf16_t* __restrict__ colrow, int hi) {
  return *(const v16bf*)(colrow + hi * 16);
}

// ---------------------------------------------------------------------------
// Kernel 0: bulk f32 -> bf16 conversion (memory bound, one pass over operands)
// ---------------------------------------------------------------------------
__global__ void f32_to_bf16_kernel(const float* __restrict__ src,
                                   bf16_t* __restrict__ dst, int n8) {
  const int i = blockIdx.x * blockDim.x + threadIdx.x;
  if (i < n8) {
    const float* p = src + (size_t)i * 8;
    f32x4 a = *(const f32x4*)(p);
    f32x4 b = *(const f32x4*)(p + 4);
    *(v8bf*)(dst + (size_t)i * 8) = cvt8(a, b);
  }
}

// ---------------------------------------------------------------------------
// Kernel 1: fused Q/K/V projection, all-bf16 operands.
// One wave computes a 32(M) x 64(N) tile: 8 WMMAs per 32-wide k-step.
// (No explicit double-buffer: 64 acc VGPRs + 48 fragment VGPRs stays spill-free;
//  cross-iteration latency is covered by wave-level parallelism.)
// Q scaled by 1/sqrt(dk); V written TRANSPOSED as [B,H,DK,S].
// ---------------------------------------------------------------------------
__global__ void mha_qkv_gemm_kernel(const bf16_t* __restrict__ xq,
                                    const bf16_t* __restrict__ xk,
                                    const bf16_t* __restrict__ xv,
                                    const bf16_t* __restrict__ wq,
                                    const bf16_t* __restrict__ wk,
                                    const bf16_t* __restrict__ wv,
                                    const float* __restrict__ bq,
                                    const float* __restrict__ bk,
                                    const float* __restrict__ bv,
                                    bf16_t* __restrict__ qh,
                                    bf16_t* __restrict__ kh,
                                    bf16_t* __restrict__ vhT) {
  const int wave = blockIdx.x * (blockDim.x >> 5) + (threadIdx.x >> 5);
  const int lane = threadIdx.x & 31;
  const int row  = lane & 15;
  const int hi   = lane >> 4;

  const int tilesN = D_MODEL / 64;              // 16
  const int tilesM = MROWS / 32;                // 128
  const int tilesPerMat = tilesM * tilesN;      // 2048

  const int mat = wave / tilesPerMat;
  const int t   = wave % tilesPerMat;
  const int tm  = t / tilesN;
  const int tn  = t % tilesN;

  const bf16_t* X    = (mat == 0) ? xq : (mat == 1) ? xk : xv;
  const bf16_t* W    = (mat == 0) ? wq : (mat == 1) ? wk : wv;
  const float*  bias = (mat == 0) ? bq : (mat == 1) ? bk : bv;
  const float scale  = (mat == 0) ? 0.125f : 1.0f;   // 1/sqrt(64)

  const int m0 = tm * 32;
  const int n0 = tn * 64;

  v8f acc[2][4] = {{v8f{}, v8f{}, v8f{}, v8f{}}, {v8f{}, v8f{}, v8f{}, v8f{}}};

  const bf16_t* arow0 = X + (size_t)(m0 + row) * D_MODEL;
  const bf16_t* arow1 = arow0 + (size_t)16 * D_MODEL;
  const bf16_t* wrow  = W + (size_t)(n0 + row) * D_MODEL;

  for (int k0 = 0; k0 < D_MODEL; k0 += 32) {
    if (k0 + 32 < D_MODEL) {
      __builtin_prefetch(arow0 + k0 + 32, 0, 3);
      __builtin_prefetch(arow1 + k0 + 32, 0, 3);
    }
    v16bf a0 = load_a_bf16(arow0 + k0, hi);
    v16bf a1 = load_a_bf16(arow1 + k0, hi);
    v16bf bfr[4];
#pragma unroll
    for (int c = 0; c < 4; ++c)
      bfr[c] = load_b_bf16(wrow + (size_t)c * 16 * D_MODEL + k0, hi);
#pragma unroll
    for (int c = 0; c < 4; ++c) {
      acc[0][c] = __builtin_amdgcn_wmma_f32_16x16x32_bf16(
          false, a0, false, bfr[c], (short)0, acc[0][c], false, false);
      acc[1][c] = __builtin_amdgcn_wmma_f32_16x16x32_bf16(
          false, a1, false, bfr[c], (short)0, acc[1][c], false, false);
    }
  }

  // Epilogue: bias, scale, scatter.
#pragma unroll
  for (int c = 0; c < 4; ++c) {
    const int n  = n0 + c * 16 + row;             // output feature for this lane
    const int hd = n >> 6;                        // head
    const int d  = n & 63;                        // dim within head
    const float bv_ = bias[n];
#pragma unroll
    for (int sub = 0; sub < 2; ++sub) {
#pragma unroll
      for (int r8 = 0; r8 < 8; ++r8) {
        const int m  = m0 + sub * 16 + r8 + hi * 8;   // global token row
        const int bb = m >> 10;                       // batch
        const int ss = m & 1023;                      // sequence pos
        const float val = (acc[sub][c][r8] + bv_) * scale;
        const bf16_t bfv = (bf16_t)val;
        if (mat == 2) {
          // V transposed: [B, H, DK, S]
          vhT[(((size_t)(bb * NHEADS + hd) * DK + d) * SEQ) + ss] = bfv;
        } else {
          bf16_t* Y = (mat == 0) ? qh : kh;           // [B, H, S, DK]
          Y[(((size_t)(bb * NHEADS + hd) * SEQ + ss) * DK) + d] = bfv;
        }
      }
    }
  }
}

// ---------------------------------------------------------------------------
// Kernel 2: flash attention. One wave per (b, h, 16-query tile).
// K and V fragments issued together at block top so V latency hides behind
// score WMMAs + softmax. P converted C-layout -> A-layout through LDS.
// ---------------------------------------------------------------------------
__global__ void mha_flash_attn_kernel(const bf16_t* __restrict__ qh,
                                      const bf16_t* __restrict__ kh,
                                      const bf16_t* __restrict__ vhT,
                                      const int* __restrict__ mask,
                                      bf16_t* __restrict__ concat) {
  __shared__ bf16_t pbuf[16][36];                 // 16x32 P tile, padded

  const int wid  = blockIdx.x;                    // B*H*(S/16) = 4096
  const int qt   = wid & 63;                      // query tile
  const int hd   = (wid >> 6) & (NHEADS - 1);
  const int bb   = wid >> 10;
  const int lane = threadIdx.x & 31;
  const int row  = lane & 15;
  const int hi   = lane >> 4;

  const size_t headBase = (size_t)(bb * NHEADS + hd) * SEQ * DK;

  // Resident Q fragment: 16x64 split into two 16x32 A-fragments.
  const bf16_t* qrow = qh + headBase + (size_t)(qt * 16 + row) * DK;
  v16bf qa[2];
#pragma unroll
  for (int kc = 0; kc < 2; ++kc) qa[kc] = load_a_bf16(qrow + kc * 32, hi);

  float m_i[8], l_i[8];
#pragma unroll
  for (int r8 = 0; r8 < 8; ++r8) { m_i[r8] = -INFINITY; l_i[r8] = 0.0f; }
  v8f o[4] = {v8f{}, v8f{}, v8f{}, v8f{}};

  for (int j0 = 0; j0 < SEQ; j0 += 32) {
    // ---- issue ALL loads for this key block up front ----
    v16bf kb[2][2];
#pragma unroll
    for (int tt = 0; tt < 2; ++tt) {
      const bf16_t* krow = kh + headBase + (size_t)(j0 + tt * 16 + row) * DK;
#pragma unroll
      for (int kc = 0; kc < 2; ++kc)
        kb[tt][kc] = *(const v16bf*)(krow + kc * 32 + hi * 16);
    }
    v16bf vf[4];
#pragma unroll
    for (int c = 0; c < 4; ++c) {
      const bf16_t* vrow = vhT + ((size_t)(bb * NHEADS + hd) * DK + c * 16 + row) * SEQ
                         + j0 + hi * 16;
      vf[c] = *(const v16bf*)(vrow);
    }
    const int mv0 = mask[bb * SEQ + j0 + row];
    const int mv1 = mask[bb * SEQ + j0 + 16 + row];

    // ---- scores: 16x32 = Q(16x64) . K(32 keys x 64)^T ----
    v8f s[2] = {v8f{}, v8f{}};
#pragma unroll
    for (int tt = 0; tt < 2; ++tt)
#pragma unroll
      for (int kc = 0; kc < 2; ++kc)
        s[tt] = __builtin_amdgcn_wmma_f32_16x16x32_bf16(
            false, qa[kc], false, kb[tt][kc], (short)0, s[tt], false, false);

    // ---- mask (per key column) ----
#pragma unroll
    for (int r8 = 0; r8 < 8; ++r8) {
      s[0][r8] = (mv0 != 0) ? s[0][r8] : -1e9f;
      s[1][r8] = (mv1 != 0) ? s[1][r8] : -1e9f;
    }

    // ---- online softmax (row = r8 + 8*hi; its 16 cols live in one 16-lane group) ----
#pragma unroll
    for (int r8 = 0; r8 < 8; ++r8) {
      float bm = fmaxf(s[0][r8], s[1][r8]);
#pragma unroll
      for (int off = 1; off < 16; off <<= 1)
        bm = fmaxf(bm, __shfl_xor(bm, off, 16));
      const float mnew = fmaxf(m_i[r8], bm);
      const float corr = __expf(m_i[r8] - mnew);
      const float p0 = __expf(s[0][r8] - mnew);
      const float p1 = __expf(s[1][r8] - mnew);
      float rs = p0 + p1;
#pragma unroll
      for (int off = 1; off < 16; off <<= 1)
        rs += __shfl_xor(rs, off, 16);
      l_i[r8] = l_i[r8] * corr + rs;
      m_i[r8] = mnew;
#pragma unroll
      for (int c = 0; c < 4; ++c) o[c][r8] *= corr;
      // stash P into LDS (C layout -> row-major)
      pbuf[r8 + 8 * hi][0 * 16 + row] = (bf16_t)p0;
      pbuf[r8 + 8 * hi][1 * 16 + row] = (bf16_t)p1;
    }
    __syncthreads();

    // ---- reload P in A layout (16x32) ----
    v16bf pf;
#pragma unroll
    for (int i = 0; i < 8; ++i) {
      pf[i]     = pbuf[row][hi * 16 + i];
      pf[i + 8] = pbuf[row][hi * 16 + 8 + i];
    }
    __syncthreads();

    // ---- O += P . V ----
#pragma unroll
    for (int c = 0; c < 4; ++c)
      o[c] = __builtin_amdgcn_wmma_f32_16x16x32_bf16(
          false, pf, false, vf[c], (short)0, o[c], false, false);
  }

  // ---- epilogue: normalize, write concat [B, S, H*DK] as bf16 ----
#pragma unroll
  for (int r8 = 0; r8 < 8; ++r8) {
    const float inv = 1.0f / l_i[r8];
    const int ss = qt * 16 + r8 + 8 * hi;
#pragma unroll
    for (int c = 0; c < 4; ++c) {
      const int feat = hd * DK + c * 16 + row;
      concat[((size_t)(bb * SEQ + ss) * D_MODEL) + feat] = (bf16_t)(o[c][r8] * inv);
    }
  }
}

// ---------------------------------------------------------------------------
// Kernel 3: output projection  out = concat @ Wo^T + bo  (f32 out).
// One wave computes a 32(M) x 64(N) tile, all-bf16 operands.
// ---------------------------------------------------------------------------
__global__ void mha_out_proj_kernel(const bf16_t* __restrict__ concat,
                                    const bf16_t* __restrict__ wo,
                                    const float* __restrict__ bo,
                                    float* __restrict__ out) {
  const int wave = blockIdx.x * (blockDim.x >> 5) + (threadIdx.x >> 5);
  const int lane = threadIdx.x & 31;
  const int row  = lane & 15;
  const int hi   = lane >> 4;

  const int tilesN = D_MODEL / 64;              // 16
  const int tm = wave / tilesN;
  const int tn = wave % tilesN;
  const int m0 = tm * 32;
  const int n0 = tn * 64;

  v8f acc[2][4] = {{v8f{}, v8f{}, v8f{}, v8f{}}, {v8f{}, v8f{}, v8f{}, v8f{}}};
  const bf16_t* arow0 = concat + (size_t)(m0 + row) * D_MODEL;
  const bf16_t* arow1 = arow0 + (size_t)16 * D_MODEL;
  const bf16_t* wrow  = wo + (size_t)(n0 + row) * D_MODEL;

  for (int k0 = 0; k0 < D_MODEL; k0 += 32) {
    if (k0 + 32 < D_MODEL) {
      __builtin_prefetch(arow0 + k0 + 32, 0, 3);
      __builtin_prefetch(arow1 + k0 + 32, 0, 3);
    }
    v16bf a0 = load_a_bf16(arow0 + k0, hi);
    v16bf a1 = load_a_bf16(arow1 + k0, hi);
    v16bf bfr[4];
#pragma unroll
    for (int c = 0; c < 4; ++c)
      bfr[c] = load_b_bf16(wrow + (size_t)c * 16 * D_MODEL + k0, hi);
#pragma unroll
    for (int c = 0; c < 4; ++c) {
      acc[0][c] = __builtin_amdgcn_wmma_f32_16x16x32_bf16(
          false, a0, false, bfr[c], (short)0, acc[0][c], false, false);
      acc[1][c] = __builtin_amdgcn_wmma_f32_16x16x32_bf16(
          false, a1, false, bfr[c], (short)0, acc[1][c], false, false);
    }
  }

#pragma unroll
  for (int c = 0; c < 4; ++c) {
    const int n = n0 + c * 16 + row;
    const float bv_ = bo[n];
#pragma unroll
    for (int sub = 0; sub < 2; ++sub) {
#pragma unroll
      for (int r8 = 0; r8 < 8; ++r8) {
        const int m = m0 + sub * 16 + r8 + hi * 8;
        out[(size_t)m * D_MODEL + n] = acc[sub][c][r8] + bv_;
      }
    }
  }
}

// ---------------------------------------------------------------------------
extern "C" void kernel_launch(void* const* d_in, const int* in_sizes, int n_in,
                              void* d_out, int out_size, void* d_ws, size_t ws_size,
                              hipStream_t stream) {
  (void)in_sizes; (void)n_in; (void)out_size; (void)ws_size;

  const float* q    = (const float*)d_in[0];
  const float* k    = (const float*)d_in[1];
  const float* v    = (const float*)d_in[2];
  const int*   mask = (const int*)  d_in[3];
  const float* Wq   = (const float*)d_in[4];
  const float* bq   = (const float*)d_in[5];
  const float* Wk   = (const float*)d_in[6];
  const float* bk   = (const float*)d_in[7];
  const float* Wv   = (const float*)d_in[8];
  const float* bv   = (const float*)d_in[9];
  const float* Wo   = (const float*)d_in[10];
  const float* bo   = (const float*)d_in[11];
  float* out = (float*)d_out;

  // Workspace layout (bf16 elements):
  //   [0]      wq,wk,wv,wo : 4 x 1M
  //   [4M]     xq,xk,xv    : 3 x 4M   (xq region reused as `concat` later)
  //   [16M]    qh,kh,vhT   : 3 x 4M
  const size_t MW = (size_t)D_MODEL * D_MODEL;  // 1M
  const size_t MX = (size_t)MROWS * D_MODEL;    // 4M
  bf16_t* ws  = (bf16_t*)d_ws;
  bf16_t* wqb = ws + 0 * MW;
  bf16_t* wkb = ws + 1 * MW;
  bf16_t* wvb = ws + 2 * MW;
  bf16_t* wob = ws + 3 * MW;
  bf16_t* xq  = ws + 4 * MW;
  bf16_t* xk  = xq + MX;
  bf16_t* xv  = xk + MX;
  bf16_t* qh  = xv + MX;
  bf16_t* kh  = qh + MX;
  bf16_t* vhT = kh + MX;
  bf16_t* concat = xq;                          // alias: tokens dead after kernel 1

  // Kernel 0: convert operands to bf16.
  {
    const int n8w = (int)(MW / 8);
    const int n8x = (int)(MX / 8);
    f32_to_bf16_kernel<<<(n8x + 255) / 256, 256, 0, stream>>>(q,  xq,  n8x);
    f32_to_bf16_kernel<<<(n8x + 255) / 256, 256, 0, stream>>>(k,  xk,  n8x);
    f32_to_bf16_kernel<<<(n8x + 255) / 256, 256, 0, stream>>>(v,  xv,  n8x);
    f32_to_bf16_kernel<<<(n8w + 255) / 256, 256, 0, stream>>>(Wq, wqb, n8w);
    f32_to_bf16_kernel<<<(n8w + 255) / 256, 256, 0, stream>>>(Wk, wkb, n8w);
    f32_to_bf16_kernel<<<(n8w + 255) / 256, 256, 0, stream>>>(Wv, wvb, n8w);
    f32_to_bf16_kernel<<<(n8w + 255) / 256, 256, 0, stream>>>(Wo, wob, n8w);
  }

  // Kernel 1: 3 matrices x 2048 wave-tiles, 4 waves per 128-thread block.
  {
    const int waves = 3 * (MROWS / 32) * (D_MODEL / 64);   // 6144
    mha_qkv_gemm_kernel<<<waves / 4, 128, 0, stream>>>(
        xq, xk, xv, wqb, wkb, wvb, bq, bk, bv, qh, kh, vhT);
  }

  // Kernel 2: one wave per (b, h, 16-row query tile) = 4096 blocks of 32.
  {
    const int blocks = BATCH * NHEADS * (SEQ / 16);        // 4096
    mha_flash_attn_kernel<<<blocks, 32, 0, stream>>>(qh, kh, vhT, mask, concat);
  }

  // Kernel 3: 2048 wave-tiles, 4 waves per block.
  {
    const int waves = (MROWS / 32) * (D_MODEL / 64);       // 2048
    mha_out_proj_kernel<<<waves / 4, 128, 0, stream>>>(concat, wob, bo, out);
  }
}
